// GATClassifier_33346126086714
// MI455X (gfx1250) — matrix operand order repaired and checked by reading, not compiled, not run on hardware
//
#include <hip/hip_runtime.h>
#include <math.h>

#define N_NODES 100000
#define N_EDGES 1600000
#define EN (N_EDGES + N_NODES)   // edges + self loops = 1,700,000
#define F_IN 128
#define HD 64
#define NH 4
#define DH 16
#define NG 128
#define NC 10
#define BN_EPS 1e-5f
#define NEG_SLOPE 0.2f

typedef __attribute__((ext_vector_type(16))) _Float16 v16h;
typedef __attribute__((ext_vector_type(8)))  float    v8f;

// ---------------------------------------------------------------- utilities

__global__ __launch_bounds__(256) void k_init_f(float* __restrict__ p, float v, int n) {
  int i = blockIdx.x * 256 + threadIdx.x;
  if (i < n) p[i] = v;
}

// f32 -> f16 weight pre-conversion (runs once per call over a few KB)
__global__ __launch_bounds__(256) void k_cvt_f16(const float* __restrict__ src,
                                                 _Float16* __restrict__ dst, int n) {
  int i = blockIdx.x * 256 + threadIdx.x;
  if (i < n) dst[i] = (_Float16)src[i];
}

__device__ __forceinline__ void edge_sd(const int* __restrict__ ei, int e, int& s, int& d) {
  if (e < N_EDGES) { s = ei[e]; d = ei[N_EDGES + e]; }
  else             { s = e - N_EDGES; d = s; }          // self loop
}

__device__ __forceinline__ float leaky(float x) {
  return x > 0.f ? x : NEG_SLOPE * x;
}

// float atomic max via int/uint punning (init must be -INF)
__device__ __forceinline__ void atomicMaxF(float* addr, float v) {
  if (v >= 0.f) atomicMax((int*)addr, __float_as_int(v));
  else          atomicMin((unsigned int*)addr, __float_as_uint(v));
}

// ------------------------------------------------------- WMMA GEMM (f16 in, f32 acc)
// C[Nrows,64] = A[Nrows,K] @ Wh[K,64] (Wh already f16).  Nrows % 32 == 0, K templated.
// 8 waves / block: wave>>2 selects row tile (2 per block), wave&3 selects col tile (4 total).
// All K/32 B-tiles are hoisted into registers; K-loop fully unrolled -> back-to-back WMMAs.
template <int K>
__global__ __launch_bounds__(256) void k_gemm_wmma(const float* __restrict__ A,
                                                   const _Float16* __restrict__ Wh,
                                                   float* __restrict__ C,
                                                   int Nrows) {
  constexpr int KSTEPS = K / 32;
  const int wave = threadIdx.x >> 5;
  const int lane = threadIdx.x & 31;
  const int rowBase = (blockIdx.x * 2 + (wave >> 2)) * 16;
  const int colBase = (wave & 3) * 16;
  if (rowBase >= Nrows) return;                 // wave-uniform: EXEC stays all-1s
  const int l16 = lane & 15;
  const int hi  = lane >> 4;                    // half-wave select

  const float* __restrict__ Arow = A + (size_t)(rowBase + l16) * K;
  const int aKbase = hi * 8;                    // A: ISA 16-bit layout (K 0-7/16-23 vs 8-15/24-31)
  const int bKbase = hi * 8;                    // B: same layout as A (K 0-7/16-23 vs 8-15/24-31)

  // hoist all B tiles for this column strip into registers (weights are tiny, L2-hot)
  v16h btile[KSTEPS];
#pragma unroll
  for (int kt = 0; kt < KSTEPS; ++kt) {
#pragma unroll
    for (int i = 0; i < 16; ++i) {
      int kk = kt * 32 + bKbase + i + ((i < 8) ? 0 : 8);
      btile[kt][i] = Wh[(size_t)kk * HD + colBase + l16];
    }
  }

  v8f acc = {};
#pragma unroll
  for (int kt = 0; kt < KSTEPS; ++kt) {
    v16h a;
#pragma unroll
    for (int i = 0; i < 16; ++i) {
      int kk = kt * 32 + aKbase + i + ((i < 8) ? 0 : 8);   // ISA 16-bit A layout
      a[i] = (_Float16)Arow[kk];
    }
    acc = __builtin_amdgcn_wmma_f32_16x16x32_f16(
        /*neg_a=*/false, a, /*neg_b=*/false, btile[kt],
        /*c_mod=*/(short)0, acc, /*reuse_a=*/false, /*reuse_b=*/false);
  }

#pragma unroll
  for (int r = 0; r < 8; ++r) {
    int row = rowBase + r + hi * 8;             // C/D layout: M = r + 8*hi, N = l16
    C[(size_t)row * HD + colBase + l16] = acc[r];
  }
}

// ------------------------------------------- per-node attention coefficients
__global__ __launch_bounds__(256) void k_alpha(const float* __restrict__ h,
                                               const float* __restrict__ a_src,
                                               const float* __restrict__ a_dst,
                                               float* __restrict__ as_,
                                               float* __restrict__ ad_) {
  int gid = blockIdx.x * 256 + threadIdx.x;
  if (gid >= N_NODES * NH) return;
  int n = gid >> 2, hh = gid & 3;
  const float* hp = h + (size_t)n * HD + hh * DH;
  const float* sp = a_src + hh * DH;
  const float* dp = a_dst + hh * DH;
  float s = 0.f, d = 0.f;
#pragma unroll
  for (int k = 0; k < DH; ++k) { s += hp[k] * sp[k]; d += hp[k] * dp[k]; }
  as_[gid] = s;
  ad_[gid] = d;
}

// ------------------------------------------------------------- edge passes
__global__ __launch_bounds__(256) void k_edge_max(const int* __restrict__ ei,
                                                  const float* __restrict__ as_,
                                                  const float* __restrict__ ad_,
                                                  float* __restrict__ m) {
  int gid = blockIdx.x * 256 + threadIdx.x;
  if (gid >= EN * NH) return;
  int e = gid >> 2, hh = gid & 3;
  int s, d; edge_sd(ei, e, s, d);
  float ev = leaky(as_[s * NH + hh] + ad_[d * NH + hh]);
  atomicMaxF(&m[d * NH + hh], ev);
}

__global__ __launch_bounds__(256) void k_edge_sum(const int* __restrict__ ei,
                                                  const float* __restrict__ as_,
                                                  const float* __restrict__ ad_,
                                                  const float* __restrict__ m,
                                                  float* __restrict__ ssum) {
  int gid = blockIdx.x * 256 + threadIdx.x;
  if (gid >= EN * NH) return;
  int e = gid >> 2, hh = gid & 3;
  int s, d; edge_sd(ei, e, s, d);
  float ev = leaky(as_[s * NH + hh] + ad_[d * NH + hh]);
  float p  = expf(ev - m[d * NH + hh]);
  atomicAdd(&ssum[d * NH + hh], p);
}

// one thread per (edge, feature): lanes cover contiguous features -> coalesced h[src] gather
__global__ __launch_bounds__(256) void k_edge_agg(const int* __restrict__ ei,
                                                  const float* __restrict__ h,
                                                  const float* __restrict__ as_,
                                                  const float* __restrict__ ad_,
                                                  const float* __restrict__ m,
                                                  const float* __restrict__ ssum,
                                                  float* __restrict__ agg) {
  int gid = blockIdx.x * 256 + threadIdx.x;
  if (gid >= EN * HD) return;
  int e = gid >> 6, f = gid & 63, hh = f >> 4;
  int s, d; edge_sd(ei, e, s, d);
  float ev    = leaky(as_[s * NH + hh] + ad_[d * NH + hh]);
  float p     = expf(ev - m[d * NH + hh]);
  float alpha = p / (ssum[d * NH + hh] + 1e-16f);
  float val   = h[(size_t)s * HD + f] * alpha;
  atomicAdd(&agg[(size_t)d * HD + f], val);
}

// -------------------------------------------------- bias + BN(eval) + ReLU
__global__ __launch_bounds__(256) void k_bn_relu(const float* __restrict__ agg,
                                                 const float* __restrict__ bias,
                                                 const float* __restrict__ gamma,
                                                 const float* __restrict__ beta,
                                                 float* __restrict__ act) {
  int gid = blockIdx.x * 256 + threadIdx.x;
  if (gid >= N_NODES * HD) return;
  int f = gid & 63;
  float scale = gamma[f] * rsqrtf(1.f + BN_EPS);
  float v = (agg[gid] + bias[f]) * scale + beta[f];
  act[gid] = v > 0.f ? v : 0.f;
}

// ---------------------------------------------------------- global mean pool
__global__ __launch_bounds__(256) void k_pool(const float* __restrict__ act,
                                              const int* __restrict__ batch,
                                              float* __restrict__ pooled,
                                              float* __restrict__ counts) {
  int gid = blockIdx.x * 256 + threadIdx.x;
  if (gid >= N_NODES * HD) return;
  int n = gid >> 6, f = gid & 63;
  int g = batch[n];
  atomicAdd(&pooled[g * HD + f], act[gid]);
  if (f == 0) atomicAdd(&counts[g], 1.f);
}

// -------------------------------------------- classifier head + log_softmax
__global__ __launch_bounds__(128) void k_head(const float* __restrict__ pooled,
                                              const float* __restrict__ counts,
                                              const float* __restrict__ Wf1,
                                              const float* __restrict__ bf1,
                                              const float* __restrict__ g3,
                                              const float* __restrict__ be3,
                                              const float* __restrict__ Wf2,
                                              const float* __restrict__ bf2,
                                              float* __restrict__ out) {
  int g = threadIdx.x;
  if (g >= NG) return;
  float cnt = counts[g]; if (cnt < 1.f) cnt = 1.f;
  float p[HD];
#pragma unroll
  for (int k = 0; k < HD; ++k) p[k] = pooled[g * HD + k] / cnt;

  float z[HD];
  float scale = rsqrtf(1.f + BN_EPS);
  for (int j = 0; j < HD; ++j) {
    float acc = bf1[j];
    for (int k = 0; k < HD; ++k) acc += p[k] * Wf1[k * HD + j];
    acc = acc * (g3[j] * scale) + be3[j];
    z[j] = acc > 0.f ? acc : 0.f;
  }

  float logit[NC];
  float mx = -INFINITY;
  for (int c = 0; c < NC; ++c) {
    float acc = bf2[c];
    for (int j = 0; j < HD; ++j) acc += z[j] * Wf2[j * NC + c];
    logit[c] = acc;
    mx = fmaxf(mx, acc);
  }
  float sum = 0.f;
  for (int c = 0; c < NC; ++c) sum += expf(logit[c] - mx);
  float lse = mx + logf(sum);
  for (int c = 0; c < NC; ++c) out[g * NC + c] = logit[c] - lse;
}

// ---------------------------------------------------------------- dispatcher

extern "C" void kernel_launch(void* const* d_in, const int* in_sizes, int n_in,
                              void* d_out, int out_size, void* d_ws, size_t ws_size,
                              hipStream_t stream) {
  const float* x      = (const float*)d_in[0];
  const int*   ei     = (const int*)  d_in[1];
  const int*   batch  = (const int*)  d_in[2];
  const float* W1     = (const float*)d_in[3];
  const float* aS1    = (const float*)d_in[4];
  const float* aD1    = (const float*)d_in[5];
  const float* b1     = (const float*)d_in[6];
  const float* W2     = (const float*)d_in[7];
  const float* aS2    = (const float*)d_in[8];
  const float* aD2    = (const float*)d_in[9];
  const float* b2     = (const float*)d_in[10];
  const float* g1     = (const float*)d_in[11];
  const float* be1    = (const float*)d_in[12];
  const float* g2     = (const float*)d_in[13];
  const float* be2    = (const float*)d_in[14];
  const float* Wf1    = (const float*)d_in[15];
  const float* bf1    = (const float*)d_in[16];
  const float* g3     = (const float*)d_in[17];
  const float* be3    = (const float*)d_in[18];
  const float* Wf2    = (const float*)d_in[19];
  const float* bf2    = (const float*)d_in[20];
  float* out = (float*)d_out;

  // workspace layout (floats, then f16 weights)
  float* ws = (float*)d_ws;
  const size_t N64 = (size_t)N_NODES * HD;   // 6.4e6
  const size_t N4  = (size_t)N_NODES * NH;   // 4e5
  float* hbuf   = ws;                // [N,64] transformed features
  float* actbuf = hbuf   + N64;      // [N,64] post BN+ReLU
  float* aggbuf = actbuf + N64;      // [N,64] aggregation target
  float* as_    = aggbuf + N64;      // [N,4]
  float* ad_    = as_    + N4;       // [N,4]
  float* mbuf   = ad_    + N4;       // [N,4]
  float* sbuf   = mbuf   + N4;       // [N,4]
  float* pooled = sbuf   + N4;       // [G,64]
  float* counts = pooled + (size_t)NG * HD;  // [G]
  _Float16* W1h = (_Float16*)(counts + NG);  // [128,64] f16
  _Float16* W2h = W1h + F_IN * HD;           // [64,64]  f16

  const int BLK = 256;
  const int gN4   = (int)((N4 + BLK - 1) / BLK);
  const int gN64  = (int)((N64 + BLK - 1) / BLK);
  const int gE4   = (EN * NH + BLK - 1) / BLK;
  const int gE64  = (int)(((size_t)EN * HD + BLK - 1) / BLK);
  const int gGemm = N_NODES / 32;    // 3125, exact

  // weight pre-conversion (f32 -> f16), once per call
  k_cvt_f16<<<(F_IN * HD + BLK - 1) / BLK, BLK, 0, stream>>>(W1, W1h, F_IN * HD);
  k_cvt_f16<<<(HD * HD + BLK - 1) / BLK, BLK, 0, stream>>>(W2, W2h, HD * HD);

  // ---------------- GAT layer 1 (K = 128) ----------------
  k_gemm_wmma<F_IN><<<gGemm, BLK, 0, stream>>>(x, W1h, hbuf, N_NODES);
  k_alpha<<<gN4, BLK, 0, stream>>>(hbuf, aS1, aD1, as_, ad_);
  k_init_f<<<gN4,  BLK, 0, stream>>>(mbuf, -INFINITY, (int)N4);
  k_init_f<<<gN4,  BLK, 0, stream>>>(sbuf, 0.f, (int)N4);
  k_init_f<<<gN64, BLK, 0, stream>>>(aggbuf, 0.f, (int)N64);
  k_edge_max<<<gE4, BLK, 0, stream>>>(ei, as_, ad_, mbuf);
  k_edge_sum<<<gE4, BLK, 0, stream>>>(ei, as_, ad_, mbuf, sbuf);
  k_edge_agg<<<gE64, BLK, 0, stream>>>(ei, hbuf, as_, ad_, mbuf, sbuf, aggbuf);
  k_bn_relu<<<gN64, BLK, 0, stream>>>(aggbuf, b1, g1, be1, actbuf);

  // ---------------- GAT layer 2 (K = 64) ----------------
  k_gemm_wmma<HD><<<gGemm, BLK, 0, stream>>>(actbuf, W2h, hbuf, N_NODES);
  k_alpha<<<gN4, BLK, 0, stream>>>(hbuf, aS2, aD2, as_, ad_);
  k_init_f<<<gN4,  BLK, 0, stream>>>(mbuf, -INFINITY, (int)N4);
  k_init_f<<<gN4,  BLK, 0, stream>>>(sbuf, 0.f, (int)N4);
  k_init_f<<<gN64, BLK, 0, stream>>>(aggbuf, 0.f, (int)N64);
  k_edge_max<<<gE4, BLK, 0, stream>>>(ei, as_, ad_, mbuf);
  k_edge_sum<<<gE4, BLK, 0, stream>>>(ei, as_, ad_, mbuf, sbuf);
  k_edge_agg<<<gE64, BLK, 0, stream>>>(ei, hbuf, as_, ad_, mbuf, sbuf, aggbuf);
  k_bn_relu<<<gN64, BLK, 0, stream>>>(aggbuf, b2, g2, be2, actbuf);

  // ---------------- pool + classifier ----------------
  k_init_f<<<(NG * HD + NG + BLK - 1) / BLK, BLK, 0, stream>>>(pooled, 0.f, NG * HD + NG);
  k_pool<<<gN64, BLK, 0, stream>>>(actbuf, batch, pooled, counts);
  k_head<<<1, 128, 0, stream>>>(pooled, counts, Wf1, bf1, g3, be3, Wf2, bf2, out);
}
